// HeteroGATConv_43645457662486
// MI455X (gfx1250) — matrix operand order repaired and checked by reading, not compiled, not run on hardware
//
#include <hip/hip_runtime.h>
#include <hip/hip_bf16.h>

// ---------------------------------------------------------------------------
// HeteroGATConv on MI455X (gfx1250):
//   dense phase : bf16 WMMA GEMMs (v_wmma_f32_16x16x32_bf16, fp32 accum)
//                 - B panel staged per-workgroup into LDS via
//                   global_load_async_to_lds_b128 (ASYNCcnt) then ds_load
//                 - register-blocked 16x64 per wave (A fragment reused x4)
//   edge phase  : wave-per-edge logits + scatter softmax with f32 atomics
//
// Workspace layout (bytes, 256-aligned chunks), N=50000, C=256, E=200000:
//   xb16[3]   : 3 * N*C * 2   (bf16 features)            ~76.8 MB
//   Wp[8]     : 8 * 65536 * 2 (packed bf16 W fragments)   ~1.0 MB
//   xl[4]     : 4 * N*C * 4   (lin_l outputs, fp32)     ~204.8 MB
//   xr[4]     : 4 * N*C * 4   (lin_r outputs, fp32)     ~204.8 MB
//   wlog[4]   : 4 * E * 4     (logits -> exp weights)     ~3.2 MB
//   mmono[4]  : 4 * N * 4     (monotonic-uint seg max)    ~0.8 MB
//   ssum[4]   : 4 * N * 4     (softmax denominators)      ~0.8 MB
// ---------------------------------------------------------------------------

typedef __attribute__((ext_vector_type(16))) __bf16 v16bf;
typedef __attribute__((ext_vector_type(8)))  __bf16 v8bf;
typedef __attribute__((ext_vector_type(8)))  float  v8f;

__device__ __forceinline__ __bf16 f2bf(float f) {
  union { float f; unsigned u; } x; x.f = f;
  unsigned r = (x.u + 0x7fffu + ((x.u >> 16) & 1u)) >> 16;
  union { unsigned short s; __bf16 b; } y; y.s = (unsigned short)r;
  return y.b;
}

// order-preserving float -> uint for atomicMax-based segment max
__device__ __forceinline__ unsigned fmono(float f) {
  unsigned u = __float_as_uint(f);
  return (u & 0x80000000u) ? ~u : (u | 0x80000000u);
}
__device__ __forceinline__ float fmono_inv(unsigned u) {
  unsigned b = (u & 0x80000000u) ? (u & 0x7fffffffu) : ~u;
  return __uint_as_float(b);
}

// --------------------------- fp32 -> bf16 convert ---------------------------
__global__ void k_to_bf16(const float* __restrict__ src, __bf16* __restrict__ dst,
                          long n) {
  long i = (long)blockIdx.x * 256 + threadIdx.x;
  if (i < n) dst[i] = f2bf(src[i]);
}

// ------------------- pack W (row-major KxN fp32) into WMMA B fragments ------
// Packed element order: [tileN(16)][kb(8)][lane(32)][j(16)]
//   lane holds column n = tileN*16 + (lane&15)
//   K = kb*32 + ((j&8)?16:0) + ((lane&16)?8:0) + (j&7)   (16-bit B: matches A table order)
__global__ void k_pack_w(const float* __restrict__ W, __bf16* __restrict__ Wp) {
  int idx  = blockIdx.x * 256 + threadIdx.x;          // 65536 total
  int j    = idx & 15;
  int lane = (idx >> 4) & 31;
  int kb   = (idx >> 9) & 7;
  int tn   = idx >> 12;
  int n = tn * 16 + (lane & 15);
  int k = kb * 32 + ((j & 8) ? 16 : 0) + ((lane & 16) ? 8 : 0) + (j & 7);
  Wp[idx] = f2bf(W[k * 256 + n]);
}

// --------------------------- WMMA GEMM: Y = X*W + b -------------------------
// X: [Mrows,256] bf16 row-major.  Wp: packed fragments.  Y: [Mrows,256] fp32.
// Block = 8 waves. The block's B panel (4 N-tiles x K=256, 32 KB) is staged
// into LDS with global_load_async_to_lds_b128, then each wave computes a
// 16x64 strip: one A fragment feeds 4 WMMAs per k-step, B read via ds_load.
__global__ __launch_bounds__(256)
void k_gemm_bf16(const __bf16* __restrict__ X, const __bf16* __restrict__ Wp,
                 const float* __restrict__ bias, float* __restrict__ Y,
                 int Mrows) {
  __shared__ __attribute__((aligned(32))) __bf16 sB[4 * 8 * 32 * 16]; // 32 KB
  const int lane   = threadIdx.x & 31;
  const int tileN4 = blockIdx.y;                       // group of 4 N-tiles
  const int Mtiles = (Mrows + 15) >> 4;
  int tileM = blockIdx.x * 8 + (threadIdx.x >> 5);
  if (tileM >= Mtiles) tileM = Mtiles - 1;   // duplicate work; keep barrier+EXEC

  // ---- async-stage packed B panel into LDS (ASYNCcnt path) ----
  {
    const char* gsrc = (const char*)(Wp + (size_t)(tileN4 * 4) * (8 * 32 * 16));
    unsigned ldsbase = (unsigned)(uintptr_t)(&sB[0]);   // flat low 32b = LDS off
#pragma unroll
    for (int it = 0; it < 16; ++it) {
      unsigned chunk = ((unsigned)it * 256u + threadIdx.x) * 16u;  // bytes
      unsigned           ldst  = ldsbase + chunk;
      unsigned long long gaddr = (unsigned long long)gsrc + chunk;
      asm volatile("global_load_async_to_lds_b128 %0, %1, off"
                   :: "v"(ldst), "v"(gaddr) : "memory");
    }
    asm volatile("s_wait_asynccnt 0" ::: "memory");
  }
  __syncthreads();

  int mrow = tileM * 16 + (lane & 15);
  if (mrow >= Mrows) mrow = Mrows - 1;                 // clamp (padding rows)
  const __bf16* aRow  = X + (size_t)mrow * 256;
  const int     kOffA = (lane & 16) ? 8 : 0;           // 16-bit A layout
  const __bf16* sBl   = sB + (size_t)lane * 16;        // per-lane B base

  v8f acc[4] = {};
#pragma unroll
  for (int kb = 0; kb < 8; ++kb) {
    const int kbase = kb * 32;
    v8bf a0 = *(const v8bf*)(aRow + kbase + kOffA);       // K = base+off..+7
    v8bf a1 = *(const v8bf*)(aRow + kbase + 16 + kOffA);  // K = base+16+off..
    v16bf a = __builtin_shufflevector(a0, a1,
        0,1,2,3,4,5,6,7,8,9,10,11,12,13,14,15);
#pragma unroll
    for (int t = 0; t < 4; ++t) {
      v16bf b = *(const v16bf*)(sBl + (size_t)((t * 8 + kb) * 32) * 16);
      acc[t] = __builtin_amdgcn_wmma_f32_16x16x32_bf16(
          false, a, false, b, (short)0, acc[t], false, false);
    }
  }

  const int r0 = tileM * 16 + ((lane & 16) ? 8 : 0);     // C/D layout
#pragma unroll
  for (int t = 0; t < 4; ++t) {
    const int   col = (tileN4 * 4 + t) * 16 + (lane & 15);
    const float bc  = bias[col];
#pragma unroll
    for (int r = 0; r < 8; ++r) {
      int row = r0 + r;
      if (row < Mrows) Y[(size_t)row * 256 + col] = acc[t][r] + bc;
    }
  }
}

// ----------------------- edge logits + segment max --------------------------
// wave per edge; lane covers 8 channels; shuffle-reduce; lane0 writes + atomics
__global__ __launch_bounds__(256)
void k_edge_logits(const int* __restrict__ ei, const float* __restrict__ xl,
                   const float* __restrict__ xr, const float* __restrict__ att,
                   float* __restrict__ elog, unsigned* __restrict__ mmono,
                   int E) {
  const int lane = threadIdx.x & 31;
  const int e = blockIdx.x * 8 + (threadIdx.x >> 5);
  if (e >= E) return;
  const int src = ei[e], dst = ei[E + e];
  const float4* pl = (const float4*)(xl + (size_t)src * 256) + lane * 2;
  const float4* pr = (const float4*)(xr + (size_t)dst * 256) + lane * 2;
  const float4* pa = (const float4*)att + lane * 2;
  float acc = 0.f;
#pragma unroll
  for (int q = 0; q < 2; ++q) {
    float4 l = pl[q], r = pr[q], a = pa[q];
    float h;
    h = l.x + r.x; acc += (h > 0.f ? h : 0.2f * h) * a.x;
    h = l.y + r.y; acc += (h > 0.f ? h : 0.2f * h) * a.y;
    h = l.z + r.z; acc += (h > 0.f ? h : 0.2f * h) * a.z;
    h = l.w + r.w; acc += (h > 0.f ? h : 0.2f * h) * a.w;
  }
#pragma unroll
  for (int m = 16; m > 0; m >>= 1) acc += __shfl_xor(acc, m, 32);
  if (lane == 0) {
    elog[e] = acc;
    atomicMax(mmono + dst, fmono(acc));
  }
}

// -------------------------- exp + segment sum -------------------------------
__global__ void k_edge_wexp(const int* __restrict__ ei, float* __restrict__ elog,
                            const unsigned* __restrict__ mmono,
                            float* __restrict__ ssum, int E) {
  int e = blockIdx.x * 256 + threadIdx.x;
  if (e >= E) return;
  int dst = ei[E + e];
  float m = fmono_inv(mmono[dst]);
  float w = __expf(elog[e] - m);
  elog[e] = w;
  atomicAdd(ssum + dst, w);
}

// --------------------- weighted scatter into output -------------------------
// block per edge, thread per channel
__global__ __launch_bounds__(256)
void k_edge_scatter(const int* __restrict__ ei, const float* __restrict__ w,
                    const float* __restrict__ ssum, const float* __restrict__ xl,
                    float* __restrict__ out, int E) {
  int e = blockIdx.x;
  int src = ei[e], dst = ei[E + e];
  float alpha = w[e] / (ssum[dst] + 1e-16f);
  int c = threadIdx.x;
  atomicAdd(out + (size_t)dst * 256 + c, alpha * xl[(size_t)src * 256 + c]);
}

// ------------------------- output bias init ---------------------------------
// out = [out_A | out_B | out_C];  out_A<-b_2, out_B<-b_0, out_C<-b_1+b_3
__global__ void k_init_out(float* __restrict__ out,
                           const float* __restrict__ b2,
                           const float* __restrict__ b0,
                           const float* __restrict__ b1,
                           const float* __restrict__ b3, long NC) {
  long i = (long)blockIdx.x * 256 + threadIdx.x;
  if (i >= 3 * NC) return;
  int c = (int)(i & 255);
  float v;
  if (i < NC)          v = b2[c];
  else if (i < 2 * NC) v = b0[c];
  else                 v = b1[c] + b3[c];
  out[i] = v;
}

// ---------------------------------------------------------------------------
extern "C" void kernel_launch(void* const* d_in, const int* in_sizes, int n_in,
                              void* d_out, int out_size, void* d_ws, size_t ws_size,
                              hipStream_t stream) {
  const int  Cc = 256;
  const int  N  = in_sizes[0] / Cc;
  const int  E  = in_sizes[3] / 2;
  const long NC = (long)N * Cc;

  const float* x[3] = {(const float*)d_in[0], (const float*)d_in[1],
                       (const float*)d_in[2]};
  const int* ei[4];
  for (int r = 0; r < 4; ++r) ei[r] = (const int*)d_in[3 + r];
  const float *Wl[4], *bl[4], *Wr[4], *br[4], *att[4], *bb[4];
  for (int i = 0; i < 4; ++i) {
    int base = 7 + i * 6;
    Wl[i]  = (const float*)d_in[base + 0];
    bl[i]  = (const float*)d_in[base + 1];
    Wr[i]  = (const float*)d_in[base + 2];
    br[i]  = (const float*)d_in[base + 3];
    att[i] = (const float*)d_in[base + 4];
    bb[i]  = (const float*)d_in[base + 5];
  }

  // ---- carve workspace ----
  char* p = (char*)d_ws;
  auto take = [&](size_t bytes) {
    char* q = p;
    p += (bytes + 255) & ~(size_t)255;
    return q;
  };
  __bf16* xb[3];
  for (int t = 0; t < 3; ++t) xb[t] = (__bf16*)take(NC * 2);
  __bf16* wp[8];
  for (int g = 0; g < 8; ++g) wp[g] = (__bf16*)take(65536 * 2);
  float* xl[4]; float* xr[4];
  for (int r = 0; r < 4; ++r) xl[r] = (float*)take(NC * 4);
  for (int r = 0; r < 4; ++r) xr[r] = (float*)take(NC * 4);
  float* wlog[4]; unsigned* mmono[4]; float* ssum[4];
  for (int r = 0; r < 4; ++r) wlog[r]  = (float*)take((size_t)E * 4);
  for (int r = 0; r < 4; ++r) mmono[r] = (unsigned*)take((size_t)N * 4);
  for (int r = 0; r < 4; ++r) ssum[r]  = (float*)take((size_t)N * 4);

  float* out = (float*)d_out;
  const int srcT[4] = {0, 1, 2, 0};   // A->B, B->C, C->A, A->C
  const int dstT[4] = {1, 2, 0, 2};

  // ---- 1: feature convert to bf16 ----
  int cvB = (int)((NC + 255) / 256);
  for (int t = 0; t < 3; ++t)
    k_to_bf16<<<cvB, 256, 0, stream>>>(x[t], xb[t], NC);

  // ---- 2: pack weights into WMMA B-fragment order ----
  for (int i = 0; i < 4; ++i) {
    k_pack_w<<<256, 256, 0, stream>>>(Wl[i], wp[2 * i]);
    k_pack_w<<<256, 256, 0, stream>>>(Wr[i], wp[2 * i + 1]);
  }

  // ---- 3: WMMA GEMMs (xl_r = x_src@Wl+bl, xr_r = x_dst@Wr+br) ----
  int Mtiles = (N + 15) / 16;
  dim3 gg((Mtiles + 7) / 8, 4);
  for (int r = 0; r < 4; ++r) {
    k_gemm_bf16<<<gg, 256, 0, stream>>>(xb[srcT[r]], wp[2 * r],     bl[r], xl[r], N);
    k_gemm_bf16<<<gg, 256, 0, stream>>>(xb[dstT[r]], wp[2 * r + 1], br[r], xr[r], N);
  }

  // ---- 4: zero softmax state, init output with fused biases ----
  for (int r = 0; r < 4; ++r) {
    hipMemsetAsync(mmono[r], 0, (size_t)N * 4, stream);
    hipMemsetAsync(ssum[r],  0, (size_t)N * 4, stream);
  }
  int ioB = (int)((3 * NC + 255) / 256);
  k_init_out<<<ioB, 256, 0, stream>>>(out, bb[2], bb[0], bb[1], bb[3], NC);

  // ---- 5: edge phase per relation ----
  for (int r = 0; r < 4; ++r) {
    float* outp = out + (long)dstT[r] * NC;
    k_edge_logits<<<(E + 7) / 8, 256, 0, stream>>>(ei[r], xl[r], xr[r], att[r],
                                                   wlog[r], mmono[r], E);
    k_edge_wexp<<<(E + 255) / 256, 256, 0, stream>>>(ei[r], wlog[r], mmono[r],
                                                     ssum[r], E);
    k_edge_scatter<<<E, 256, 0, stream>>>(ei[r], wlog[r], ssum[r], xl[r],
                                          outp, E);
  }
}
